// SakeInteractionBlock_10033043603848
// MI455X (gfx1250) — matrix operand order repaired and checked by reading, not compiled, not run on hardware
//
#include <hip/hip_runtime.h>
#include <hip/hip_bf16.h>

// ---------------------------------------------------------------------------
// Types
// ---------------------------------------------------------------------------
typedef __bf16 bh_t;
typedef bh_t  v16bf __attribute__((ext_vector_type(16)));
typedef float v8f   __attribute__((ext_vector_type(8)));

// Fragment table: all weights packed as bf16 B-fragments (512 halves each).
// frag layout: for fragment (kt,nt): half index = lane*16 + e,
//   N = nt*16 + (lane&15),  K = kt*32 + (e&8)*2 + (lane>>4)*8 + (e&7)
#define OFF_WIN    0    // K=128 (KT4),  N=43->48  (NT3)  : 12
#define OFF_WOUT1  12   // K=172->192(6),N=64 (4)         : 24
#define OFF_WOUT2  36   // K=64 (2),     N=64 (4)         : 8
#define OFF_SEM    44   // K=64 (2),     N=7->16 (1)      : 2
#define OFF_MIX    46   // K=448 (14),   N=448 (28)       : 392
#define OFF_POST1  438  // K=448 (14),   N=64 (4)         : 56
#define OFF_POST2  494  // K=64 (2),     N=64 (4)         : 8
#define OFF_NODE1  502  // K=576 (18),   N=64 (4)         : 72
#define OFF_NODE2  574  // K=64 (2),     N=64 (4)         : 8
#define FRAG_TOTAL 582

__device__ const int FR_START[10] = {0,12,36,44,46,438,494,502,574,582};
__device__ const int FR_NT[9]     = {3,4,4,1,28,4,4,4,4};
__device__ const int FR_KD[9]     = {128,172,64,64,448,448,64,576,64};
__device__ const int FR_ND[9]     = {43,64,64,7,448,64,64,64,64};

// ---------------------------------------------------------------------------
// Helpers
// ---------------------------------------------------------------------------
// Native f32 -> bf16 convert (lowers to v_cvt_*bf16* on gfx1250; far fewer
// VALU ops than a manual RNE sequence).
__device__ __forceinline__ unsigned short f2bfu(float f) {
  union { bh_t b; unsigned short s; } u;
  u.b = (bh_t)f;
  return u.s;
}

// Two converts packed into one dword (compiler can fuse to v_cvt_pk_bf16_f32).
__device__ __forceinline__ unsigned pack2bf(float lo, float hi) {
  union { bh_t b[2]; unsigned u; } x;
  x.b[0] = (bh_t)lo;
  x.b[1] = (bh_t)hi;
  return x.u;
}

__device__ __forceinline__ void wave_fence() {
  // same-wave LDS write->read ordering on CDNA5 (split counters, wave32)
  asm volatile("s_wait_dscnt 0" ::: "memory");
}

__device__ __forceinline__ v8f wmma_bf16(v16bf a, v16bf b, v8f c) {
  return __builtin_amdgcn_wmma_f32_16x16x32_bf16(false, a, false, b, (short)0, c,
                                                 false, false);
}

// Load A fragment (16x32 bf16) from an LDS tile of bf16 rows (stride in halves).
// ISA layout: lane L: row = L&15, g = L>>4; VGPR v holds K = kt*32 + (v&4)*4 +
// (v&3)*2 + {0,1} + g*8  (pairs are consecutive K -> one b32 LDS read each).
__device__ __forceinline__ v16bf load_a(const unsigned short* tile, int stride, int kt) {
  int lane = threadIdx.x & 31;
  int row  = lane & 15;
  int g    = (lane >> 4) & 1;
  union { v16bf v; unsigned u[8]; } a;
  #pragma unroll
  for (int v = 0; v < 8; ++v) {
    int kb = kt * 32 + ((v & 4) << 2) + ((v & 3) << 1) + (g << 3);
    a.u[v] = *(const unsigned*)(tile + row * stride + kb);
  }
  return a.v;
}

// Load a pre-packed B fragment: 16 contiguous halves per lane (32B).
__device__ __forceinline__ v16bf load_b(const unsigned short* pk, int frag) {
  int lane = threadIdx.x & 31;
  const uint4* p = (const uint4*)(pk + (((size_t)frag) << 9) + (lane << 4));
  union { v16bf v; uint4 q[2]; } b;
  b.q[0] = p[0];
  b.q[1] = p[1];
  return b.v;
}

__device__ __forceinline__ float silu_f(float x) { return x / (1.f + __expf(-x)); }

__device__ __forceinline__ void atomicMaxF(float* addr, float val) {
  if (val >= 0.f) atomicMax((int*)addr, __float_as_int(val));
  else            atomicMin((unsigned int*)addr, __float_as_uint(val));
}

// ---------------------------------------------------------------------------
// Weight packing (f32 -> bf16 B-fragment layout)
// ---------------------------------------------------------------------------
__global__ void pack_weights_kernel(unsigned short* pk,
    const float* w0, const float* w1, const float* w2, const float* w3,
    const float* w4, const float* w5, const float* w6, const float* w7,
    const float* w8) {
  const float* wsrc[9] = {w0, w1, w2, w3, w4, w5, w6, w7, w8};
  int total = FRAG_TOTAL * 512;
  for (int i = blockIdx.x * blockDim.x + threadIdx.x; i < total;
       i += gridDim.x * blockDim.x) {
    int frag = i >> 9, within = i & 511;
    int s = 0;
    while (s < 8 && frag >= FR_START[s + 1]) ++s;
    int lf = frag - FR_START[s];
    int nt = lf % FR_NT[s], kt = lf / FR_NT[s];
    int lane = within >> 4, e = within & 15;
    int n = nt * 16 + (lane & 15);
    int k = kt * 32 + ((e & 8) << 1) + ((lane >> 4) << 3) + (e & 7);
    float v = (k < FR_KD[s] && n < FR_ND[s]) ? wsrc[s][k * FR_ND[s] + n] : 0.f;
    pk[i] = f2bfu(v);
  }
}

// ---------------------------------------------------------------------------
// Edge pipeline: qcat -> mlp_in -> rbf filter -> mlp_out -> sem logits
// One wave per 16-edge tile, 8 waves/block.
// ---------------------------------------------------------------------------
__global__ __launch_bounds__(256) void edge_kernel(
    const float* __restrict__ q, const float* __restrict__ d_ij,
    const int* __restrict__ idx_i, const int* __restrict__ idx_j,
    const float* __restrict__ rbf_off, const float* __restrict__ rbf_w,
    const unsigned short* __restrict__ pk,
    const float* __restrict__ win_b, const float* __restrict__ wout1_b,
    const float* __restrict__ wout2_b, const float* __restrict__ sem_b,
    float* __restrict__ qij_out, float* __restrict__ att_out,
    float* __restrict__ amax, int P) {
  __shared__ unsigned short hA[8][16 * 192];
  __shared__ float dsh[8][16];
  __shared__ int   jsh[8][16];

  int w    = threadIdx.x >> 5;
  int lane = threadIdx.x & 31;
  int tile = blockIdx.x * 8 + w;
  if (tile * 16 >= P) return;
  int row = lane & 15, g = lane >> 4;
  unsigned short* A = &hA[w][0];

  int e  = tile * 16 + row;
  int ec = e < P ? e : P - 1;
  int src  = g ? idx_j[ec] : idx_i[ec];
  float dv = d_ij[ec];
  if (g == 0) { dsh[w][row] = dv; jsh[w][row] = idx_j[ec]; }

  // qcat tile: lane fills cols [g*64, g*64+64) of its row, bf16 (b64 stores)
  const float4* qr = (const float4*)(q + (size_t)src * 64);
  #pragma unroll
  for (int j4 = 0; j4 < 16; ++j4) {
    float4 v = qr[j4];
    int c = g * 64 + j4 * 4;
    uint2 pr;
    pr.x = pack2bf(v.x, v.y);
    pr.y = pack2bf(v.z, v.w);
    *(uint2*)&A[row * 192 + c] = pr;
  }
  if (g == 0) {
    for (int c = 172; c < 192; ++c) A[row * 192 + c] = 0;  // zero pad
    A[row * 192 + 171] = f2bfu(dv);                        // d_ij column
  }
  wave_fence();

  // GEMM1: qcat[16x128] x win_w -> q_lin; apply RBF filter into cols 128..170
  #pragma unroll
  for (int nt = 0; nt < 3; ++nt) {
    v8f acc = {};
    #pragma unroll
    for (int kt = 0; kt < 4; ++kt)
      acc = wmma_bf16(load_a(A, 192, kt), load_b(pk, OFF_WIN + kt * 3 + nt), acc);
    int col = nt * 16 + (lane & 15);
    if (col < 43) {
      float bias = win_b[col];
      float off = rbf_off[col], wid = rbf_w[col];
      #pragma unroll
      for (int r = 0; r < 8; ++r) {
        int rr = r + g * 8;
        float t = (dsh[w][rr] - off) / wid;
        float filt = __expf(-0.5f * t * t) * (acc[r] + bias);
        A[rr * 192 + 128 + col] = f2bfu(filt);
      }
    }
  }
  wave_fence();

  // GEMM2: h[16x192] x wout1 -> silu -> 64
  v8f h2[4];
  #pragma unroll
  for (int nt = 0; nt < 4; ++nt) {
    v8f acc = {};
    #pragma unroll
    for (int kt = 0; kt < 6; ++kt)
      acc = wmma_bf16(load_a(A, 192, kt), load_b(pk, OFF_WOUT1 + kt * 4 + nt), acc);
    float bias = wout1_b[nt * 16 + (lane & 15)];
    #pragma unroll
    for (int r = 0; r < 8; ++r) acc[r] = silu_f(acc[r] + bias);
    h2[nt] = acc;
  }
  wave_fence();
  #pragma unroll
  for (int nt = 0; nt < 4; ++nt) {
    int col = nt * 16 + (lane & 15);
    #pragma unroll
    for (int r = 0; r < 8; ++r) A[(r + g * 8) * 192 + col] = f2bfu(h2[nt][r]);
  }
  wave_fence();

  // GEMM3: -> q_ij_mtx [16x64], store f32 + stage bf16 for sem GEMM
  #pragma unroll
  for (int nt = 0; nt < 4; ++nt) {
    v8f acc = {};
    #pragma unroll
    for (int kt = 0; kt < 2; ++kt)
      acc = wmma_bf16(load_a(A, 192, kt), load_b(pk, OFF_WOUT2 + kt * 4 + nt), acc);
    int col = nt * 16 + (lane & 15);
    float bias = wout2_b[col];
    #pragma unroll
    for (int r = 0; r < 8; ++r) {
      float x = acc[r] + bias;
      int erow = tile * 16 + r + g * 8;
      if (erow < P) qij_out[(size_t)erow * 64 + col] = x;
      h2[nt][r] = x;
    }
  }
  wave_fence();
  #pragma unroll
  for (int nt = 0; nt < 4; ++nt) {
    int col = nt * 16 + (lane & 15);
    #pragma unroll
    for (int r = 0; r < 8; ++r) A[(r + g * 8) * 192 + col] = f2bfu(h2[nt][r]);
  }
  wave_fence();

  // sem GEMM + CELU(alpha=2) + segment max
  {
    v8f acc = {};
    #pragma unroll
    for (int kt = 0; kt < 2; ++kt)
      acc = wmma_bf16(load_a(A, 192, kt), load_b(pk, OFF_SEM + kt), acc);
    int col = lane & 15;
    if (col < 7) {
      float bias = sem_b[col];
      #pragma unroll
      for (int r = 0; r < 8; ++r) {
        int rr = r + g * 8;
        int erow = tile * 16 + rr;
        float x = acc[r] + bias;
        float cel = x > 0.f ? x : 2.f * (__expf(0.5f * x) - 1.f);
        if (erow < P) {
          att_out[(size_t)erow * 7 + col] = cel;
          atomicMaxF(&amax[(size_t)jsh[w][rr] * 7 + col], cel);
        }
      }
    }
  }
}

// ---------------------------------------------------------------------------
// Scatter softmax (3 passes)
// ---------------------------------------------------------------------------
__global__ void softmax1_kernel(const float* __restrict__ att,
                                const int* __restrict__ idx_j,
                                const float* __restrict__ amax,
                                float* __restrict__ ebuf, float* __restrict__ denom,
                                float* __restrict__ cnt, int P) {
  int t = blockIdx.x * blockDim.x + threadIdx.x;
  if (t >= P * 7) return;
  int e = t / 7, h = t - e * 7;
  int j = idx_j[e];
  float ex = __expf(att[t] - amax[(size_t)j * 7 + h]);
  ebuf[t] = ex;
  atomicAdd(&denom[(size_t)j * 7 + h], ex);
  if (h == 0) atomicAdd(&cnt[j], 1.f);
}

__global__ void softmax2_kernel(const int* __restrict__ idx_j,
                                const float* __restrict__ denom,
                                float* __restrict__ ebuf, float* __restrict__ agg, int P) {
  int t = blockIdx.x * blockDim.x + threadIdx.x;
  if (t >= P * 7) return;
  int e = t / 7, h = t - e * 7;
  int j = idx_j[e];
  float c = ebuf[t] / denom[(size_t)j * 7 + h];
  ebuf[t] = c;
  atomicAdd(&agg[(size_t)j * 7 + h], c);
}

__global__ void softmax3_kernel(const int* __restrict__ idx_j,
                                const float* __restrict__ agg,
                                float* __restrict__ ebuf, int P) {
  int t = blockIdx.x * blockDim.x + threadIdx.x;
  if (t >= P * 7) return;
  int e = t / 7, h = t - e * 7;
  ebuf[t] = ebuf[t] / agg[(size_t)idx_j[e] * 7 + h];
}

// ---------------------------------------------------------------------------
// Mix GEMM [16x448]x[448x448] + tanh + scatter (csum, q_agg)
// 8 waves cooperate on one 16-edge tile (A shared in LDS).
// ---------------------------------------------------------------------------
__global__ __launch_bounds__(256) void mix_kernel(
    const float* __restrict__ qij, const float* __restrict__ cmbG,
    const float* __restrict__ r_ij, const float* __restrict__ d_ij,
    const int* __restrict__ idx_jG, const unsigned short* __restrict__ pk,
    float* __restrict__ q_agg, float* __restrict__ csum, int P) {
  __shared__ unsigned short Am[16 * 448];
  __shared__ float qm[16 * 64];
  __shared__ float cb[16 * 8];
  __shared__ float rj[16 * 4];
  __shared__ int   jj[16];

  int t  = threadIdx.x;
  int e0 = blockIdx.x * 16;
  if (e0 >= P) return;
  int rv = P - e0 < 16 ? P - e0 : 16;

  ((float4*)qm)[t] = ((const float4*)(qij + (size_t)e0 * 64))[t];
  if (t < 112) cb[(t / 7) * 8 + (t % 7)] = cmbG[(size_t)e0 * 7 + t];
  if (t < 16) {
    int ee = e0 + t < P ? e0 + t : P - 1;
    float dv = d_ij[ee] + 1e-5f;
    jj[t] = idx_jG[ee];
    rj[t * 4 + 0] = r_ij[(size_t)ee * 3 + 0] / dv;
    rj[t * 4 + 1] = r_ij[(size_t)ee * 3 + 1] / dv;
    rj[t * 4 + 2] = r_ij[(size_t)ee * 3 + 2] / dv;
  }
  __syncthreads();

  // build q_ij_att tile (2 elements/thread/iter -> one b32 DS store)
  for (int i = t * 2; i < 16 * 448; i += 512) {
    int row = i / 448, c = i - row * 448;
    int f0 = c / 7,        nh0 = c - f0 * 7;
    int f1 = (c + 1) / 7,  nh1 = (c + 1) - f1 * 7;
    float v0 = qm[row * 64 + f0] * cb[row * 8 + nh0];
    float v1 = qm[row * 64 + f1] * cb[row * 8 + nh1];
    if (row < rv) {
      atomicAdd(&q_agg[(size_t)jj[row] * 448 + c],     v0);
      atomicAdd(&q_agg[(size_t)jj[row] * 448 + c + 1], v1);
    }
    *(unsigned*)&Am[i] = pack2bf(v0, v1);
  }
  __syncthreads();

  int w = t >> 5, lane = t & 31, g = lane >> 4;
  for (int nt = w; nt < 28; nt += 8) {
    v8f acc = {};
    #pragma unroll
    for (int kt = 0; kt < 14; ++kt) {
      if (kt + 1 < 14)
        __builtin_prefetch(pk + (((size_t)(OFF_MIX + (kt + 1) * 28 + nt)) << 9), 0, 0);
      acc = wmma_bf16(load_a(Am, 448, kt), load_b(pk, OFF_MIX + kt * 28 + nt), acc);
    }
    int col = nt * 16 + (lane & 15);
    #pragma unroll
    for (int r = 0; r < 8; ++r) {
      int rr = r + g * 8;
      if (rr < rv) {
        float cf = tanhf(acc[r]);
        float* dst = &csum[((size_t)jj[rr] * 448 + col) * 3];
        atomicAdd(dst + 0, cf * rj[rr * 4 + 0]);
        atomicAdd(dst + 1, cf * rj[rr * 4 + 1]);
        atomicAdd(dst + 2, cf * rj[rr * 4 + 2]);
      }
    }
  }
}

// ---------------------------------------------------------------------------
// Node model: cnorm -> post MLP -> concat -> node MLP -> residual
// One wave per 16-atom tile.
// ---------------------------------------------------------------------------
__global__ __launch_bounds__(32) void node_kernel(
    const float* __restrict__ q, const float* __restrict__ q_agg,
    const float* __restrict__ csum, const float* __restrict__ cnt,
    const unsigned short* __restrict__ pk,
    const float* __restrict__ post1_b, const float* __restrict__ post2_b,
    const float* __restrict__ node1_b, const float* __restrict__ node2_b,
    float* __restrict__ out, int n) {
  __shared__ unsigned short An[16 * 576];
  __shared__ float cw[16];

  int lane = threadIdx.x;
  int a0 = blockIdx.x * 16;
  if (a0 >= n) return;
  int g = lane >> 4, l15 = lane & 15;

  if (lane < 16) {
    int ar = a0 + lane < n ? a0 + lane : n - 1;
    cw[lane] = fmaxf(cnt[ar], 1.f);
  }
  wave_fence();

  // cnorm tile -> An cols 0..447 (2 outputs per iter -> one b32 DS store)
  for (int i = lane * 2; i < 16 * 448; i += 64) {
    int row = i / 448, c = i - row * 448;
    int ar = a0 + row < n ? a0 + row : n - 1;
    const float* s = &csum[((size_t)ar * 448 + c) * 3];
    float inv = 1.f / cw[row];
    float a0m = s[0] * inv, a1m = s[1] * inv, a2m = s[2] * inv;
    float b0m = s[3] * inv, b1m = s[4] * inv, b2m = s[5] * inv;
    *(unsigned*)&An[row * 576 + c] =
        pack2bf(a0m * a0m + a1m * a1m + a2m * a2m,
                b0m * b0m + b1m * b1m + b2m * b2m);
  }
  wave_fence();

  v8f hq[4];
  // post1 (448 -> 64), silu
  #pragma unroll
  for (int nt = 0; nt < 4; ++nt) {
    v8f acc = {};
    #pragma unroll
    for (int kt = 0; kt < 14; ++kt)
      acc = wmma_bf16(load_a(An, 576, kt), load_b(pk, OFF_POST1 + kt * 4 + nt), acc);
    float bias = post1_b[nt * 16 + l15];
    #pragma unroll
    for (int r = 0; r < 8; ++r) acc[r] = silu_f(acc[r] + bias);
    hq[nt] = acc;
  }
  wave_fence();
  #pragma unroll
  for (int nt = 0; nt < 4; ++nt)
    #pragma unroll
    for (int r = 0; r < 8; ++r)
      An[(r + g * 8) * 576 + 448 + nt * 16 + l15] = f2bfu(hq[nt][r]);
  wave_fence();

  // post2 (64 -> 64), silu -> q_comb in regs
  #pragma unroll
  for (int nt = 0; nt < 4; ++nt) {
    v8f acc = {};
    #pragma unroll
    for (int kt = 0; kt < 2; ++kt)
      acc = wmma_bf16(load_a(An + 448, 576, kt), load_b(pk, OFF_POST2 + kt * 4 + nt), acc);
    float bias = post2_b[nt * 16 + l15];
    #pragma unroll
    for (int r = 0; r < 8; ++r) acc[r] = silu_f(acc[r] + bias);
    hq[nt] = acc;
  }
  wave_fence();

  // rebuild An = concat([q, q_agg, q_comb]) as bf16
  for (int i = lane * 2; i < 16 * 64; i += 64) {
    int row = i >> 6, c = i & 63;
    int ar = a0 + row < n ? a0 + row : n - 1;
    const float* s = &q[(size_t)ar * 64 + c];
    *(unsigned*)&An[row * 576 + c] = pack2bf(s[0], s[1]);
  }
  for (int i = lane * 2; i < 16 * 448; i += 64) {
    int row = i / 448, c = i - row * 448;
    int ar = a0 + row < n ? a0 + row : n - 1;
    const float* s = &q_agg[(size_t)ar * 448 + c];
    *(unsigned*)&An[row * 576 + 64 + c] = pack2bf(s[0], s[1]);
  }
  #pragma unroll
  for (int nt = 0; nt < 4; ++nt)
    #pragma unroll
    for (int r = 0; r < 8; ++r)
      An[(r + g * 8) * 576 + 512 + nt * 16 + l15] = f2bfu(hq[nt][r]);
  wave_fence();

  // node1 (576 -> 64), silu
  #pragma unroll
  for (int nt = 0; nt < 4; ++nt) {
    v8f acc = {};
    #pragma unroll
    for (int kt = 0; kt < 18; ++kt)
      acc = wmma_bf16(load_a(An, 576, kt), load_b(pk, OFF_NODE1 + kt * 4 + nt), acc);
    float bias = node1_b[nt * 16 + l15];
    #pragma unroll
    for (int r = 0; r < 8; ++r) acc[r] = silu_f(acc[r] + bias);
    hq[nt] = acc;
  }
  wave_fence();
  #pragma unroll
  for (int nt = 0; nt < 4; ++nt)
    #pragma unroll
    for (int r = 0; r < 8; ++r)
      An[(r + g * 8) * 576 + nt * 16 + l15] = f2bfu(hq[nt][r]);
  wave_fence();

  // node2 (64 -> 64), silu, residual
  #pragma unroll
  for (int nt = 0; nt < 4; ++nt) {
    v8f acc = {};
    #pragma unroll
    for (int kt = 0; kt < 2; ++kt)
      acc = wmma_bf16(load_a(An, 576, kt), load_b(pk, OFF_NODE2 + kt * 4 + nt), acc);
    int col = nt * 16 + l15;
    float bias = node2_b[col];
    #pragma unroll
    for (int r = 0; r < 8; ++r) {
      int ar = a0 + r + g * 8;
      if (ar < n) {
        float s = silu_f(acc[r] + bias);
        out[(size_t)ar * 64 + col] = q[(size_t)ar * 64 + col] + s;
      }
    }
  }
}

// ---------------------------------------------------------------------------
// Host launcher
// ---------------------------------------------------------------------------
extern "C" void kernel_launch(void* const* d_in, const int* in_sizes, int n_in,
                              void* d_out, int out_size, void* d_ws, size_t ws_size,
                              hipStream_t stream) {
  const float* q        = (const float*)d_in[0];
  // d_in[1] = mu (unused by the reference forward)
  const float* r_ij     = (const float*)d_in[2];
  const float* d_ij     = (const float*)d_in[3];
  const int*   idx_i    = (const int*)d_in[4];
  const int*   idx_j    = (const int*)d_in[5];
  const float* rbf_off  = (const float*)d_in[6];
  const float* rbf_w    = (const float*)d_in[7];
  const float* win_w    = (const float*)d_in[8];
  const float* win_b    = (const float*)d_in[9];
  const float* wout1_w  = (const float*)d_in[10];
  const float* wout1_b  = (const float*)d_in[11];
  const float* wout2_w  = (const float*)d_in[12];
  const float* wout2_b  = (const float*)d_in[13];
  const float* sem_w    = (const float*)d_in[14];
  const float* sem_b    = (const float*)d_in[15];
  const float* mix_w    = (const float*)d_in[16];
  const float* post1_w  = (const float*)d_in[17];
  const float* post1_b  = (const float*)d_in[18];
  const float* post2_w  = (const float*)d_in[19];
  const float* post2_b  = (const float*)d_in[20];
  const float* node1_w  = (const float*)d_in[21];
  const float* node1_b  = (const float*)d_in[22];
  const float* node2_w  = (const float*)d_in[23];
  const float* node2_b  = (const float*)d_in[24];

  int n = in_sizes[0] / 64;   // N_ATOMS
  int P = in_sizes[3];        // N_PAIRS

  // workspace carve-up
  char* base = (char*)d_ws;
  size_t off = 0;
  unsigned short* pk = (unsigned short*)base;
  off += (size_t)FRAG_TOTAL * 512 * 2;
  off = (off + 255) & ~(size_t)255;
  float* qij  = (float*)(base + off); off += (size_t)P * 64 * 4;
  float* att  = (float*)(base + off); off += (size_t)P * 7 * 4;
  float* ebuf = (float*)(base + off); off += (size_t)P * 7 * 4;
  char* zstart = base + off;
  float* denom = (float*)(base + off); off += (size_t)n * 7 * 4;
  float* agg   = (float*)(base + off); off += (size_t)n * 7 * 4;
  float* cnt   = (float*)(base + off); off += (size_t)n * 4;
  float* qagg  = (float*)(base + off); off += (size_t)n * 448 * 4;
  float* csum  = (float*)(base + off); off += (size_t)n * 448 * 3 * 4;
  size_t zbytes = (size_t)((base + off) - zstart);
  float* amax  = (float*)(base + off); off += (size_t)n * 7 * 4;

  hipMemsetAsync(zstart, 0, zbytes, stream);
  hipMemsetAsync(amax, 0xFF, (size_t)n * 7 * 4, stream);  // -NaN pattern for max trick

  pack_weights_kernel<<<(FRAG_TOTAL * 512 + 255) / 256, 256, 0, stream>>>(
      pk, win_w, wout1_w, wout2_w, sem_w, mix_w, post1_w, post2_w, node1_w, node2_w);

  int etiles = (P + 15) / 16;
  edge_kernel<<<(etiles + 7) / 8, 256, 0, stream>>>(
      q, d_ij, idx_i, idx_j, rbf_off, rbf_w, pk,
      win_b, wout1_b, wout2_b, sem_b, qij, att, amax, P);

  int pt = P * 7;
  softmax1_kernel<<<(pt + 255) / 256, 256, 0, stream>>>(att, idx_j, amax, ebuf, denom, cnt, P);
  softmax2_kernel<<<(pt + 255) / 256, 256, 0, stream>>>(idx_j, denom, ebuf, agg, P);
  softmax3_kernel<<<(pt + 255) / 256, 256, 0, stream>>>(idx_j, agg, ebuf, P);

  mix_kernel<<<etiles, 256, 0, stream>>>(qij, ebuf, r_ij, d_ij, idx_j, pk, qagg, csum, P);

  node_kernel<<<(n + 15) / 16, 32, 0, stream>>>(
      q, qagg, csum, cnt, pk, post1_b, post2_b, node1_b, node2_b, (float*)d_out, n);
}
